// MultiLayerBiDrectionalLSTM_59966333386918
// MI455X (gfx1250) — matrix operand order repaired and checked by reading, not compiled, not run on hardware
//
#include <hip/hip_runtime.h>
#include <hip/hip_bf16.h>

#define VOCAB 32000
#define EMB   512
#define HID   512
#define LAYERS 2
#define BATCH 64
#define SEQ   512

typedef unsigned short ushort_t;
typedef __attribute__((ext_vector_type(16))) __bf16 v16bf;
typedef __attribute__((ext_vector_type(8)))  float  v8f;
typedef int v4i __attribute__((vector_size(4 * sizeof(int))));

struct B32x2 { uint4 lo, hi; };

#if defined(__AMDGCN__) && __has_builtin(__builtin_amdgcn_global_load_async_to_lds_b128) && __has_builtin(__builtin_amdgcn_s_wait_asynccnt)
#define HAVE_ASYNC_LDS 1
#endif

// ---- helpers -------------------------------------------------------------

__device__ __forceinline__ ushort_t f2bf(float f) {
    unsigned u = __float_as_uint(f);
    unsigned r = u + 0x7FFFu + ((u >> 16) & 1u);   // round to nearest even
    return (ushort_t)(r >> 16);
}

__device__ __forceinline__ float sigmoidf(float x) {
    return 1.0f / (1.0f + expf(-x));
}

// A fragment (16x32 bf16, M rows = batch). base is row-major [*, 512] bf16.
// Lane l<16: row m=l, K = {kb..kb+7, 16+kb..16+kb+7} with kb=0; lanes 16-31: kb=8.
__device__ __forceinline__ v16bf load_fragA(const ushort_t* base, int row0, int kt, int lane) {
    const ushort_t* p = base + ((row0 + (lane & 15)) << 9) + (kt << 5) + ((lane >> 4) << 3);
    B32x2 v;
    v.lo = *(const uint4*)(p);
    v.hi = *(const uint4*)(p + 16);
    return __builtin_bit_cast(v16bf, v);
}

// B fragment (32x16 bf16, N cols = gate rows of W). base row-major [*, 512].
// Lane l<16: col n=l, K = {kb..kb+7, 16+kb..16+kb+7} with kb=0; lanes 16-31: kb=8.
__device__ __forceinline__ v16bf load_fragB(const ushort_t* base, int col0, int kt, int lane) {
    const ushort_t* p = base + ((col0 + (lane & 15)) << 9) + (kt << 5) + ((lane >> 4) << 3);
    B32x2 v;
    v.lo = *(const uint4*)(p);
    v.hi = *(const uint4*)(p + 16);
    return __builtin_bit_cast(v16bf, v);
}

// Stage BATCH*HID bf16 (64 KB) from global into LDS, two buffers at once.
__device__ __forceinline__ void stage2(ushort_t* dstX, const ushort_t* srcX,
                                       ushort_t* dstH, const ushort_t* srcH,
                                       int tid) {
#ifdef HAVE_ASYNC_LDS
    #pragma unroll 4
    for (int c = tid; c < (BATCH * HID) / 8; c += 256) {
        __builtin_amdgcn_global_load_async_to_lds_b128(
            (v4i*)(srcX + c * 8), (v4i*)(dstX + c * 8), 0, 0);
        __builtin_amdgcn_global_load_async_to_lds_b128(
            (v4i*)(srcH + c * 8), (v4i*)(dstH + c * 8), 0, 0);
    }
    __builtin_amdgcn_s_wait_asynccnt(0);
#else
    for (int c = tid; c < (BATCH * HID) / 8; c += 256) {
        ((uint4*)dstX)[c] = ((const uint4*)srcX)[c];
        ((uint4*)dstH)[c] = ((const uint4*)srcH)[c];
    }
#endif
}

// ---- kernels -------------------------------------------------------------

__global__ void lstm_init_kernel(ushort_t* hcur, int* barriers) {
    int i = blockIdx.x * 256 + threadIdx.x;
    if (i < 2 * 2 * BATCH * HID) hcur[i] = 0;
    if (i < 8) barriers[i] = 0;
}

// xe[s][b][h] = bf16(emb[x[b][s]][h])
__global__ void embed_kernel(const int* __restrict__ x,
                             const float* __restrict__ emb,
                             ushort_t* __restrict__ xe) {
    int s = blockIdx.x >> 6;       // SEQ index
    int b = blockIdx.x & 63;       // BATCH index
    int tok = x[b * SEQ + s];
    const float* src = emb + (long)tok * EMB;
    ushort_t* dst = xe + ((long)s * BATCH + b) * HID;
    for (int h = threadIdx.x; h < HID; h += 256)
        dst[h] = f2bf(src[h]);
}

// Persistent recurrent layer: 64 WGs (32 per direction), 256 threads each.
// WG (dir, ks) owns h-indices [ks*16, ks*16+16), computes all 4 gates for them.
__global__ void __launch_bounds__(256, 1)
lstm_layer_kernel(const float* __restrict__ Wx_f, const float* __restrict__ bx_f,
                  const float* __restrict__ Wh_f, const float* __restrict__ bh_f,
                  const float* __restrict__ Wx_b, const float* __restrict__ bx_b,
                  const float* __restrict__ Wh_b, const float* __restrict__ bh_b,
                  const ushort_t* __restrict__ xe,   // [SEQ][BATCH][HID] bf16 (layer-0 input)
                  ushort_t* __restrict__ hs,         // [2][SEQ][BATCH][HID] bf16 (layer-0 out)
                  ushort_t* __restrict__ hcur,       // [2 dir][2 parity][BATCH][HID] bf16
                  float* __restrict__ hfinal,        // [2][BATCH][HID] f32
                  int* __restrict__ barriers,
                  int layer) {
    const int tid  = threadIdx.x;
    const int wg   = blockIdx.x;
    const int dir  = wg >> 5;           // 0 = forward stack, 1 = backward stack
    const int ks   = wg & 31;           // h-slice index (16 wide)
    const int lane = tid & 31;
    const int wave = tid >> 5;

    const float* WxL = (dir ? Wx_b : Wx_f) + (long)layer * 4 * HID * HID;
    const float* WhL = (dir ? Wh_b : Wh_f) + (long)layer * 4 * HID * HID;
    const float* bxL = (dir ? bx_b : bx_f) + (long)layer * 4 * HID;
    const float* bhL = (dir ? bh_b : bh_f) + (long)layer * 4 * HID;

    // ---- dynamic LDS partition (282,880 B total) ----
    extern __shared__ char smem[];
    ushort_t* sWx = (ushort_t*)smem;          // [64][512] bf16 : Wx rows (g*16+kk)
    ushort_t* sWh = sWx + 64 * 512;           // [64][512] bf16
    ushort_t* sX  = sWh + 64 * 512;           // [64][512] bf16 : x_t staged
    ushort_t* sH  = sX  + 64 * 512;           // [64][512] bf16 : h_{t-1} staged
    float*    sG  = (float*)(sH + 64 * 512);  // [64 n][64 m] f32 gate pre-activations
    float*    sC  = sG + 64 * 64;             // [16 kk][64 m] f32 cell state
    float*    sB  = sC + 16 * 64;             // [64] combined bias

    // ---- preload weight slices (f32 -> bf16) ----
    for (int e = tid; e < 64 * 512; e += 256) {
        int n = e >> 9, j = e & 511;
        int g = n >> 4, kk = n & 15;
        long row = (long)g * HID + (ks * 16 + kk);
        sWx[e] = f2bf(WxL[row * HID + j]);
        sWh[e] = f2bf(WhL[row * HID + j]);
    }
    if (tid < 64) {
        int g = tid >> 4, kk = tid & 15;
        sB[tid] = bxL[g * HID + ks * 16 + kk] + bhL[g * HID + ks * 16 + kk];
    }
    for (int e = tid; e < 16 * 64; e += 256) sC[e] = 0.0f;
    __syncthreads();

    const int mt  = wave & 3;            // batch tile (M)
    const int nt0 = (wave >> 2) * 2;     // gate-col tiles (N)
    const int nt1 = nt0 + 1;

    ushort_t* hsD = hs + (long)dir * SEQ * BATCH * HID;
    int* ctr = barriers + (layer * 2 + dir);

    for (int t = 0; t < SEQ; ++t) {
        // ---- stage x_t and h_{t-1} into LDS (async global->LDS path) ----
        const ushort_t* xsrc = (layer == 0)
            ? (xe + (long)(dir ? (SEQ - 1 - t) : t) * BATCH * HID)
            : (hsD + (long)t * BATCH * HID);
        const ushort_t* hsrc = hcur + (long)(dir * 2 + (t & 1)) * BATCH * HID;
        ushort_t*       hdst = hcur + (long)(dir * 2 + ((t + 1) & 1)) * BATCH * HID;

        stage2(sX, xsrc, sH, hsrc, tid);

        if (t + 1 < SEQ) {  // prefetch next step's input (global_prefetch_b8)
            const ushort_t* xn = (layer == 0)
                ? (xe + (long)(dir ? (SEQ - 2 - t) : (t + 1)) * BATCH * HID)
                : (hsD + (long)(t + 1) * BATCH * HID);
            __builtin_prefetch(xn + tid * 128, 0, 1);
        }
        __syncthreads();

        // ---- gates[M=64, N=64] = X @ WxT + H @ WhT via bf16 WMMA ----
        v8f acc0 = {0.f,0.f,0.f,0.f,0.f,0.f,0.f,0.f};
        v8f acc1 = {0.f,0.f,0.f,0.f,0.f,0.f,0.f,0.f};
        #pragma unroll 4
        for (int kt = 0; kt < 16; ++kt) {
            v16bf a  = load_fragA(sX,  mt  * 16, kt, lane);
            v16bf b0 = load_fragB(sWx, nt0 * 16, kt, lane);
            v16bf b1 = load_fragB(sWx, nt1 * 16, kt, lane);
            acc0 = __builtin_amdgcn_wmma_f32_16x16x32_bf16(false, a, false, b0, (short)0, acc0, false, false);
            acc1 = __builtin_amdgcn_wmma_f32_16x16x32_bf16(false, a, false, b1, (short)0, acc1, false, false);
        }
        #pragma unroll 4
        for (int kt = 0; kt < 16; ++kt) {
            v16bf a  = load_fragA(sH,  mt  * 16, kt, lane);
            v16bf b0 = load_fragB(sWh, nt0 * 16, kt, lane);
            v16bf b1 = load_fragB(sWh, nt1 * 16, kt, lane);
            acc0 = __builtin_amdgcn_wmma_f32_16x16x32_bf16(false, a, false, b0, (short)0, acc0, false, false);
            acc1 = __builtin_amdgcn_wmma_f32_16x16x32_bf16(false, a, false, b1, (short)0, acc1, false, false);
        }

        // C/D layout: VGPR r, lanes 0-15 -> M=r, N=lane ; lanes 16-31 -> M=8+r, N=lane-16
        {
            int mrow = mt * 16 + ((lane >> 4) << 3);
            int nc0  = nt0 * 16 + (lane & 15);
            int nc1  = nt1 * 16 + (lane & 15);
            #pragma unroll
            for (int r = 0; r < 8; ++r) {
                sG[nc0 * 64 + mrow + r] = acc0[r];
                sG[nc1 * 64 + mrow + r] = acc1[r];
            }
        }
        __syncthreads();

        // ---- elementwise LSTM cell update (gate order f,i,g,o) ----
        for (int idx = tid; idx < 16 * 64; idx += 256) {
            int kk = idx >> 6, m = idx & 63;
            float fg = sigmoidf(sG[( 0 + kk) * 64 + m] + sB[ 0 + kk]);
            float ig = sigmoidf(sG[(16 + kk) * 64 + m] + sB[16 + kk]);
            float gg = tanhf   (sG[(32 + kk) * 64 + m] + sB[32 + kk]);
            float og = sigmoidf(sG[(48 + kk) * 64 + m] + sB[48 + kk]);
            float c  = fg * sC[idx] + ig * gg;
            sC[idx]  = c;
            float h  = og * tanhf(c);
            ushort_t hb = f2bf(h);
            int col = ks * 16 + kk;
            hdst[m * HID + col] = hb;
            if (layer == 0)
                hsD[(long)t * BATCH * HID + m * HID + col] = hb;
            if (layer == 1 && t == SEQ - 1)
                hfinal[(long)dir * BATCH * HID + m * HID + col] = h;
        }

        // ---- device-scope barrier across the 32 WGs of this direction ----
        __threadfence();
        __syncthreads();
        if (tid == 0) {
            atomicAdd(ctr, 1);
            int target = 32 * (t + 1);
            volatile int* vc = (volatile int*)ctr;
            while (*vc < target) __builtin_amdgcn_s_sleep(1);
        }
        __syncthreads();
        __threadfence();
    }
}

// out[b] = sigmoid(dot(fc_w, [h_f ; h_b]) + fc_b)
__global__ void fc_kernel(const float* __restrict__ hfinal,   // [2][BATCH][HID]
                          const float* __restrict__ fc_w,     // [1][2H]
                          const float* __restrict__ fc_b,
                          float* __restrict__ out) {
    __shared__ float red[256];
    int b = blockIdx.x;
    float acc = 0.0f;
    for (int h = threadIdx.x; h < HID; h += 256) {
        acc += hfinal[b * HID + h]               * fc_w[h];
        acc += hfinal[BATCH * HID + b * HID + h] * fc_w[HID + h];
    }
    red[threadIdx.x] = acc;
    __syncthreads();
    for (int s = 128; s > 0; s >>= 1) {
        if (threadIdx.x < s) red[threadIdx.x] += red[threadIdx.x + s];
        __syncthreads();
    }
    if (threadIdx.x == 0)
        out[b] = 1.0f / (1.0f + expf(-(red[0] + fc_b[0])));
}

// ---- host launcher -------------------------------------------------------

extern "C" void kernel_launch(void* const* d_in, const int* in_sizes, int n_in,
                              void* d_out, int out_size, void* d_ws, size_t ws_size,
                              hipStream_t stream) {
    const int*   x    = (const int*)  d_in[0];
    const float* emb  = (const float*)d_in[1];
    const float* Wx_f = (const float*)d_in[2];
    const float* bx_f = (const float*)d_in[3];
    const float* Wh_f = (const float*)d_in[4];
    const float* bh_f = (const float*)d_in[5];
    const float* Wx_b = (const float*)d_in[6];
    const float* bx_b = (const float*)d_in[7];
    const float* Wh_b = (const float*)d_in[8];
    const float* bh_b = (const float*)d_in[9];
    const float* fc_w = (const float*)d_in[10];
    const float* fc_b = (const float*)d_in[11];
    float* out = (float*)d_out;

    char* ws = (char*)d_ws;
    const size_t XE_B  = (size_t)SEQ * BATCH * HID * 2;        // 33,554,432
    const size_t HS_B  = (size_t)2 * SEQ * BATCH * HID * 2;    // 67,108,864
    const size_t HC_B  = (size_t)2 * 2 * BATCH * HID * 2;      //    262,144
    const size_t HF_B  = (size_t)2 * BATCH * HID * 4;          //    262,144
    ushort_t* xe     = (ushort_t*)(ws);
    ushort_t* hs     = (ushort_t*)(ws + XE_B);
    ushort_t* hcur   = (ushort_t*)(ws + XE_B + HS_B);
    float*    hfinal = (float*)   (ws + XE_B + HS_B + HC_B);
    int*      bar    = (int*)     (ws + XE_B + HS_B + HC_B + HF_B);

    const size_t smem = (size_t)4 * 64 * 512 * 2              // sWx,sWh,sX,sH (bf16)
                      + (size_t)(64 * 64 + 16 * 64 + 64) * 4; // sG,sC,sB (f32)
    (void)hipFuncSetAttribute((const void*)lstm_layer_kernel,
                              hipFuncAttributeMaxDynamicSharedMemorySize, (int)smem);

    lstm_init_kernel<<<512, 256, 0, stream>>>(hcur, bar);
    embed_kernel<<<SEQ * BATCH, 256, 0, stream>>>(x, emb, xe);
    lstm_layer_kernel<<<64, 256, smem, stream>>>(Wx_f, bx_f, Wh_f, bh_f,
                                                 Wx_b, bx_b, Wh_b, bh_b,
                                                 xe, hs, hcur, hfinal, bar, 0);
    lstm_layer_kernel<<<64, 256, smem, stream>>>(Wx_f, bx_f, Wh_f, bh_f,
                                                 Wx_b, bx_b, Wh_b, bh_b,
                                                 xe, hs, hcur, hfinal, bar, 1);
    fc_kernel<<<BATCH, 256, 0, stream>>>(hfinal, fc_w, fc_b, out);
}
